// VELORA_78176994722439
// MI455X (gfx1250) — compile-verified, matrix operand order checked by reading, not run on hardware
//
#include <hip/hip_runtime.h>
#include <hip/hip_bf16.h>

// ---------------- problem constants (match reference) ----------------
#define D_DIM   1024
#define M_SLOTS 128
#define SEQ     512
#define BATCH   32
#define NTOK    (BATCH * SEQ)   // 16384
#define F_DIM   (4 * D_DIM)     // 4096

typedef __bf16 bf16;
typedef unsigned int u32;
typedef __attribute__((ext_vector_type(16))) __bf16 v16bf;
typedef __attribute__((ext_vector_type(8)))  float  v8f;
typedef __attribute__((ext_vector_type(4)))  unsigned int u32x4;
typedef __attribute__((ext_vector_type(8)))  int          i32x8;
typedef __attribute__((ext_vector_type(4)))  int          i32x4;

// ---- Tensor Data Mover availability / arity detection (compile-portable) ----
#if defined(__HIP_DEVICE_COMPILE__)
#  if __has_builtin(__builtin_amdgcn_tensor_load_to_lds)
#    define VELORA_HAS_TDM 1
#  endif
#endif
#if defined(__has_include)
#  if __has_include(<hip/amd_detail/amd_gfx1250_TDM.h>)
#    define VELORA_TDM_6ARG 1
#  else
#    define VELORA_TDM_6ARG 0
#  endif
#else
#  define VELORA_TDM_6ARG 0
#endif

// ---------------- helpers ----------------
__device__ __forceinline__ float gelu_f(float x) {
    float x3 = x * x * x;
    return 0.5f * x * (1.0f + tanhf(0.7978845608f * (x + 0.044715f * x3)));
}

#ifdef VELORA_HAS_TDM
// Issue one TDM 2-D tile load: tile_h rows x tile_w bf16 elems, row stride
// `row_stride` elems, packed contiguously at LDS byte offset `lds_off`.
// D# layout per CDNA5 ISA (groups 0/1; 2-D -> groups 2/3 zero).
__device__ __forceinline__ void tdm_load_tile_2d(const bf16* gptr, u32 lds_off,
                                                 u32 tile_w, u32 tile_h,
                                                 u32 row_stride) {
    unsigned long long ga = (unsigned long long)(const void*)gptr;
    const u32 tensor_w = row_stride;        // full row length -> no OOB clip
    const u32 tensor_h = 1u << 20;
    u32x4 g0;
    g0[0] = 1u;                                            // count=1, user mode
    g0[1] = lds_off;                                       // lds_addr [63:32]
    g0[2] = (u32)(ga & 0xFFFFFFFFu);                       // global_addr lo
    g0[3] = (u32)((ga >> 32) & 0x1FFFFFFu) | (2u << 30);   // addr hi | type=2
    i32x8 g1;
    g1[0] = (int)(1u << 16);                               // data_size=1 (2B)
    g1[1] = (int)((tensor_w & 0xFFFFu) << 16);             // tensor_dim0 lo16
    g1[2] = (int)((tensor_w >> 16) & 0xFFFFu) | (int)((tensor_h & 0xFFFFu) << 16);
    g1[3] = (int)((tensor_h >> 16) & 0xFFFFu) | (int)((tile_w & 0xFFFFu) << 16);
    g1[4] = (int)(tile_h & 0xFFFFu);                       // tile_dim1; dim2=0
    g1[5] = (int)row_stride;                               // dim0_stride lo32
    g1[6] = 0;
    g1[7] = 0;
    i32x4 z4 = {0, 0, 0, 0};
#if VELORA_TDM_6ARG
    i32x8 z8 = {0, 0, 0, 0, 0, 0, 0, 0};
    __builtin_amdgcn_tensor_load_to_lds(g0, g1, z4, z4, z8, 0);
#else
    __builtin_amdgcn_tensor_load_to_lds(g0, g1, z4, z4, 0);
#endif
}
#endif // VELORA_HAS_TDM

// ---------------- conversion kernels ----------------
__global__ void velora_cvt_f32_bf16(const float* __restrict__ src,
                                    bf16* __restrict__ dst, long n) {
    long i = (long)blockIdx.x * blockDim.x + threadIdx.x;
    long stride = (long)gridDim.x * blockDim.x;
    for (; i < n; i += stride) dst[i] = (bf16)src[i];
}

// dst[c*rows + r] = src[r*cols + c]   (store W^T so B-columns are contiguous)
__global__ void velora_tcvt_f32_bf16(const float* __restrict__ src,
                                     bf16* __restrict__ dst, int rows, int cols) {
    long n = (long)rows * cols;
    long i = (long)blockIdx.x * blockDim.x + threadIdx.x;
    long stride = (long)gridDim.x * blockDim.x;
    for (; i < n; i += stride) {
        long r = i / cols;
        long c = i - r * (long)cols;
        dst[c * rows + r] = (bf16)src[i];
    }
}

// ---------------- memory attention + enhancement ----------------
__global__ void __launch_bounds__(256)
velora_mem_attn(const float* __restrict__ h,      // [N,D] f32
                const bf16*  __restrict__ h_bf,   // [N,D]
                const bf16*  __restrict__ mk,     // [M,D]
                const bf16*  __restrict__ mvt,    // [D,M]  (mem_vals^T)
                bf16*        __restrict__ enh_bf) // [N,D]
{
    __shared__ float s_scores[64 * M_SLOTS];   // 32 KB
    __shared__ bf16  s_p[64 * M_SLOTS];        // 16 KB
    const int tid   = threadIdx.x;
    const int wave  = tid >> 5;
    const int lane  = tid & 31;
    const int lrow  = lane & 15;
    const int khalf = lane >> 4;
    const int tok0  = blockIdx.x * 64;

    // ---- phase 1: scores (WMMA over K = D) ----
    {
        const int rt = wave & 3;
        const int cg = wave >> 2;
        v8f acc[4] = {};
        const bf16* arow = h_bf + (long)(tok0 + rt * 16 + lrow) * D_DIM + khalf * 16;
        for (int kk = 0; kk < D_DIM; kk += 32) {
            v16bf af = *(const v16bf*)(arow + kk);
#pragma unroll
            for (int j = 0; j < 4; ++j) {
                const bf16* brow = mk + (long)(cg * 64 + j * 16 + lrow) * D_DIM + kk + khalf * 16;
                v16bf bfr = *(const v16bf*)brow;
                acc[j] = __builtin_amdgcn_wmma_f32_16x16x32_bf16(
                    false, af, false, bfr, (short)0, acc[j], false, false);
            }
        }
#pragma unroll
        for (int j = 0; j < 4; ++j)
#pragma unroll
            for (int g = 0; g < 8; ++g) {
                int r = rt * 16 + g + 8 * khalf;
                int c = cg * 64 + j * 16 + lrow;
                s_scores[r * M_SLOTS + c] = acc[j][g] * 0.03125f;  // 1/sqrt(1024)
            }
    }
    __syncthreads();

    // ---- phase 2: row softmax over the 128 slots ----
    if (tid < 64) {
        float mx = -1e30f;
        for (int c = 0; c < M_SLOTS; ++c) mx = fmaxf(mx, s_scores[tid * M_SLOTS + c]);
        float sum = 0.0f;
        for (int c = 0; c < M_SLOTS; ++c) {
            float e = __expf(s_scores[tid * M_SLOTS + c] - mx);
            s_scores[tid * M_SLOTS + c] = e;
            sum += e;
        }
        float inv = 1.0f / sum;
        for (int c = 0; c < M_SLOTS; ++c)
            s_p[tid * M_SLOTS + c] = (bf16)(s_scores[tid * M_SLOTS + c] * inv);
    }
    __syncthreads();

    // ---- phase 3: read = P @ mem_vals (WMMA over K = M), epilogue enh ----
    {
        const int rt = wave & 3;
        const int ch = wave >> 2;
        for (int ct = 0; ct < 32; ++ct) {
            const int ncol = ch * 512 + ct * 16;
            v8f acc = {};
#pragma unroll
            for (int ks = 0; ks < M_SLOTS; ks += 32) {
                v16bf af = *(const v16bf*)(&s_p[(rt * 16 + lrow) * M_SLOTS + ks + khalf * 16]);
                v16bf bfr = *(const v16bf*)(mvt + (long)(ncol + lrow) * M_SLOTS + ks + khalf * 16);
                acc = __builtin_amdgcn_wmma_f32_16x16x32_bf16(
                    false, af, false, bfr, (short)0, acc, false, false);
            }
#pragma unroll
            for (int g = 0; g < 8; ++g) {
                int row = tok0 + rt * 16 + g + 8 * khalf;
                long idx = (long)row * D_DIM + ncol + lrow;
                enh_bf[idx] = (bf16)(h[idx] + 0.2f * acc[g]);
            }
        }
    }
}

// ---------------- masked mean pool + router softmax ----------------
__global__ void __launch_bounds__(256)
velora_pool_router(const bf16* __restrict__ enh, const int* __restrict__ mask,
                   const float* __restrict__ rw, const float* __restrict__ rb,
                   float* __restrict__ gate /* [B,2] */)
{
    __shared__ float s_pool[D_DIM];
    __shared__ float s_red[256];
    const int b = blockIdx.x;
    const int tid = threadIdx.x;

    float a0 = 0.f, a1 = 0.f, a2 = 0.f, a3 = 0.f;
    int cnt = 0;
    for (int s = 0; s < SEQ; ++s) {
        if (mask[b * SEQ + s]) {
            const bf16* row = enh + ((long)(b * SEQ + s)) * D_DIM + tid * 4;
            a0 += (float)row[0]; a1 += (float)row[1];
            a2 += (float)row[2]; a3 += (float)row[3];
            ++cnt;
        }
    }
    float denom = fmaxf((float)cnt, 1.0f);
    s_pool[tid * 4 + 0] = a0 / denom;
    s_pool[tid * 4 + 1] = a1 / denom;
    s_pool[tid * 4 + 2] = a2 / denom;
    s_pool[tid * 4 + 3] = a3 / denom;
    __syncthreads();

    float p0 = 0.f, p1 = 0.f;
    for (int d = tid; d < D_DIM; d += 256) {
        float pv = s_pool[d];
        p0 += pv * rw[d * 2 + 0];
        p1 += pv * rw[d * 2 + 1];
    }
    s_red[tid] = p0; __syncthreads();
    for (int off = 128; off > 0; off >>= 1) {
        if (tid < off) s_red[tid] += s_red[tid + off];
        __syncthreads();
    }
    float l0 = s_red[0] + rb[0];
    __syncthreads();
    s_red[tid] = p1; __syncthreads();
    for (int off = 128; off > 0; off >>= 1) {
        if (tid < off) s_red[tid] += s_red[tid + off];
        __syncthreads();
    }
    float l1 = s_red[0] + rb[1];
    if (tid == 0) {
        float mx = fmaxf(l0, l1);
        float e0 = __expf(l0 - mx), e1 = __expf(l1 - mx);
        float inv = 1.0f / (e0 + e1);
        gate[b * 2 + 0] = e0 * inv;
        gate[b * 2 + 1] = e1 * inv;
    }
}

// ---------------- 128x128x32 WMMA GEMM, TDM double-buffered staging ----------------
// C[row,n] = epilogue( A[row,:] . Wt[n,:] + bias[n] )
//   act: 0=none 1=gelu 2=relu 3=tanh
//   gate (per-batch, [B,2]) scales post-activation; gate_acc=1 -> outf +=
//   resid != null -> outf = 0.5*(val + resid)   (final residual average)
__global__ void __launch_bounds__(256)
velora_gemm(const bf16* __restrict__ A,    // [Nrows,K]
            const bf16* __restrict__ Wt,   // [Ncols,K]
            const float* __restrict__ bias,
            bf16* __restrict__ outb,
            float* __restrict__ outf,
            const float* __restrict__ gate, int gate_idx, int gate_acc,
            const float* __restrict__ resid,
            int K, int act)
{
    __shared__ bf16 sA[2][128 * 32];  // 2 x 8 KB (ping-pong)
    __shared__ bf16 sB[2][128 * 32];
    const int tid   = threadIdx.x;
    const int wave  = tid >> 5;
    const int lane  = tid & 31;
    const int lrow  = lane & 15;
    const int khalf = lane >> 4;
    const int col0  = blockIdx.x * 128;
    const int row0  = blockIdx.y * 128;
    const int Ncols = gridDim.x * 128;
    const int wr = wave >> 2;      // 0..1 -> 64-row strip
    const int wc = wave & 3;       // 0..3 -> 32-col strip

    v8f acc[4][2] = {};

#ifdef VELORA_HAS_TDM
    // ---- Tensor Data Mover pipeline: wave 0 drives DMA, all waves compute ----
    if (wave == 0) {
        tdm_load_tile_2d(A  + (long)row0 * K, (u32)(size_t)&sA[0][0], 32, 128, (u32)K);
        tdm_load_tile_2d(Wt + (long)col0 * K, (u32)(size_t)&sB[0][0], 32, 128, (u32)K);
    }
    int ib = 0;
    for (int kk = 0; kk < K; kk += 32, ib ^= 1) {
        __syncthreads();             // everyone finished reading buffer ib^1
        if (wave == 0) {
            if (kk + 32 < K) {
                tdm_load_tile_2d(A  + (long)row0 * K + (kk + 32),
                                 (u32)(size_t)&sA[ib ^ 1][0], 32, 128, (u32)K);
                tdm_load_tile_2d(Wt + (long)col0 * K + (kk + 32),
                                 (u32)(size_t)&sB[ib ^ 1][0], 32, 128, (u32)K);
                __builtin_amdgcn_s_wait_tensorcnt((short)2);  // current pair landed
            } else {
                __builtin_amdgcn_s_wait_tensorcnt((short)0);
            }
        }
        __syncthreads();             // buffer ib ready for all waves

        v16bf afr[4], bfr[2];
#pragma unroll
        for (int i = 0; i < 4; ++i)
            afr[i] = *(const v16bf*)(&sA[ib][(wr * 64 + i * 16 + lrow) * 32 + khalf * 16]);
#pragma unroll
        for (int j = 0; j < 2; ++j)
            bfr[j] = *(const v16bf*)(&sB[ib][(wc * 32 + j * 16 + lrow) * 32 + khalf * 16]);
#pragma unroll
        for (int i = 0; i < 4; ++i)
#pragma unroll
            for (int j = 0; j < 2; ++j)
                acc[i][j] = __builtin_amdgcn_wmma_f32_16x16x32_bf16(
                    false, afr[i], false, bfr[j], (short)0, acc[i][j], false, false);
    }
#else
    // ---- fallback: cooperative VALU staging (host pass / non-TDM toolchain) ----
    for (int kk = 0; kk < K; kk += 32) {
        __syncthreads();
#pragma unroll
        for (int i = 0; i < 4; ++i) {
            int c  = tid + i * 256;
            int r  = c >> 2;
            int kh = (c & 3) * 8;
            *(uint4*)(&sA[0][r * 32 + kh]) = *(const uint4*)(A  + (long)(row0 + r) * K + kk + kh);
            *(uint4*)(&sB[0][r * 32 + kh]) = *(const uint4*)(Wt + (long)(col0 + r) * K + kk + kh);
        }
        if (kk + 32 < K) {
            __builtin_prefetch(A  + (long)(row0 + (tid >> 1)) * K + kk + 32, 0, 1);
            __builtin_prefetch(Wt + (long)(col0 + (tid >> 1)) * K + kk + 32, 0, 1);
        }
        __syncthreads();
        v16bf afr[4], bfr[2];
#pragma unroll
        for (int i = 0; i < 4; ++i)
            afr[i] = *(const v16bf*)(&sA[0][(wr * 64 + i * 16 + lrow) * 32 + khalf * 16]);
#pragma unroll
        for (int j = 0; j < 2; ++j)
            bfr[j] = *(const v16bf*)(&sB[0][(wc * 32 + j * 16 + lrow) * 32 + khalf * 16]);
#pragma unroll
        for (int i = 0; i < 4; ++i)
#pragma unroll
            for (int j = 0; j < 2; ++j)
                acc[i][j] = __builtin_amdgcn_wmma_f32_16x16x32_bf16(
                    false, afr[i], false, bfr[j], (short)0, acc[i][j], false, false);
    }
#endif

    // block-uniform gate: 128-row blocks never straddle a 512-row batch
    float gval = 1.0f;
    if (gate) gval = gate[(row0 >> 9) * 2 + gate_idx];

#pragma unroll
    for (int i = 0; i < 4; ++i)
#pragma unroll
        for (int j = 0; j < 2; ++j) {
            const int n = col0 + wc * 32 + j * 16 + lrow;
            const float bs = bias ? bias[n] : 0.0f;
#pragma unroll
            for (int g = 0; g < 8; ++g) {
                const int row = row0 + wr * 64 + i * 16 + g + 8 * khalf;
                float v = acc[i][j][g] + bs;
                if (act == 1)      v = gelu_f(v);
                else if (act == 2) v = fmaxf(v, 0.0f);
                else if (act == 3) v = tanhf(v);
                v *= gval;
                const long idx = (long)row * Ncols + n;
                if (outf) {
                    if (resid)         outf[idx] = 0.5f * (v + resid[idx]);
                    else if (gate_acc) outf[idx] += v;
                    else               outf[idx] = v;
                }
                if (outb) outb[idx] = (bf16)v;
            }
        }
}

// ---------------- host-side orchestration ----------------
extern "C" void kernel_launch(void* const* d_in, const int* in_sizes, int n_in,
                              void* d_out, int out_size, void* d_ws, size_t ws_size,
                              hipStream_t stream) {
    (void)in_sizes; (void)n_in; (void)out_size; (void)ws_size;
    const float* h        = (const float*)d_in[0];
    const int*   mask     = (const int*)  d_in[1];
    const float* mem_keys = (const float*)d_in[2];
    const float* mem_vals = (const float*)d_in[3];
    const float* router_w = (const float*)d_in[4];
    const float* router_b = (const float*)d_in[5];
    const float* aw1 = (const float*)d_in[6];  const float* ab1 = (const float*)d_in[7];
    const float* aw2 = (const float*)d_in[8];  const float* ab2 = (const float*)d_in[9];
    const float* lw1 = (const float*)d_in[10]; const float* lb1 = (const float*)d_in[11];
    const float* lw2 = (const float*)d_in[12]; const float* lb2 = (const float*)d_in[13];
    const float* fw  = (const float*)d_in[14]; const float* fb  = (const float*)d_in[15];
    const float* cw1 = (const float*)d_in[16]; const float* cb1 = (const float*)d_in[17];
    const float* cw2 = (const float*)d_in[18]; const float* cb2 = (const float*)d_in[19];
    float* out = (float*)d_out;

    // workspace carve-up (256B aligned)
    char* ws = (char*)d_ws;
    size_t off = 0;
    auto carve = [&](size_t bytes) -> char* {
        char* p = ws + off;
        off = (off + bytes + 255) & ~(size_t)255;
        return p;
    };
    bf16*  h_bf     = (bf16*) carve((size_t)NTOK * D_DIM * 2);
    bf16*  enh_bf   = (bf16*) carve((size_t)NTOK * D_DIM * 2);
    bf16*  mk_bf    = (bf16*) carve((size_t)M_SLOTS * D_DIM * 2);
    bf16*  mvt_bf   = (bf16*) carve((size_t)M_SLOTS * D_DIM * 2);
    bf16*  aw1t     = (bf16*) carve((size_t)D_DIM * F_DIM * 2);
    bf16*  aw2t     = (bf16*) carve((size_t)D_DIM * F_DIM * 2);
    bf16*  lw1t     = (bf16*) carve((size_t)D_DIM * F_DIM * 2);
    bf16*  lw2t     = (bf16*) carve((size_t)D_DIM * F_DIM * 2);
    bf16*  fwt      = (bf16*) carve((size_t)D_DIM * D_DIM * 2);
    bf16*  c1t      = (bf16*) carve((size_t)D_DIM * D_DIM * 2);
    bf16*  c2t      = (bf16*) carve((size_t)D_DIM * D_DIM * 2);
    float* gate     = (float*)carve((size_t)BATCH * 2 * 4);
    bf16*  hid_bf   = (bf16*) carve((size_t)NTOK * F_DIM * 2);
    float* comb_f   = (float*)carve((size_t)NTOK * D_DIM * 4);
    bf16*  comb_bf  = (bf16*) carve((size_t)NTOK * D_DIM * 2);
    bf16*  fused_bf = (bf16*) carve((size_t)NTOK * D_DIM * 2);
    float* fused_f  = (float*)carve((size_t)NTOK * D_DIM * 4);
    bf16*  ctxh_bf  = (bf16*) carve((size_t)NTOK * D_DIM * 2);

    const dim3 blk(256);
    const dim3 cvt_grid(2048);

    // 1) precision conversion + weight transposes
    velora_cvt_f32_bf16 <<<cvt_grid, blk, 0, stream>>>(h, h_bf, (long)NTOK * D_DIM);
    velora_cvt_f32_bf16 <<<cvt_grid, blk, 0, stream>>>(mem_keys, mk_bf, (long)M_SLOTS * D_DIM);
    velora_tcvt_f32_bf16<<<cvt_grid, blk, 0, stream>>>(mem_vals, mvt_bf, M_SLOTS, D_DIM);
    velora_tcvt_f32_bf16<<<cvt_grid, blk, 0, stream>>>(aw1, aw1t, D_DIM, F_DIM);
    velora_tcvt_f32_bf16<<<cvt_grid, blk, 0, stream>>>(aw2, aw2t, F_DIM, D_DIM);
    velora_tcvt_f32_bf16<<<cvt_grid, blk, 0, stream>>>(lw1, lw1t, D_DIM, F_DIM);
    velora_tcvt_f32_bf16<<<cvt_grid, blk, 0, stream>>>(lw2, lw2t, F_DIM, D_DIM);
    velora_tcvt_f32_bf16<<<cvt_grid, blk, 0, stream>>>(fw,  fwt,  D_DIM, D_DIM);
    velora_tcvt_f32_bf16<<<cvt_grid, blk, 0, stream>>>(cw1, c1t,  D_DIM, D_DIM);
    velora_tcvt_f32_bf16<<<cvt_grid, blk, 0, stream>>>(cw2, c2t,  D_DIM, D_DIM);

    // 2) working-memory attention -> enh (bf16)
    velora_mem_attn<<<dim3(NTOK / 64), blk, 0, stream>>>(h, h_bf, mk_bf, mvt_bf, enh_bf);

    // 3) router gate
    velora_pool_router<<<dim3(BATCH), blk, 0, stream>>>(enh_bf, mask, router_w, router_b, gate);

    // 4) experts (gate-weighted combine fused into second GEMM epilogue)
    velora_gemm<<<dim3(F_DIM / 128, NTOK / 128), blk, 0, stream>>>(
        enh_bf, aw1t, ab1, hid_bf, nullptr, nullptr, 0, 0, nullptr, D_DIM, /*gelu*/1);
    velora_gemm<<<dim3(D_DIM / 128, NTOK / 128), blk, 0, stream>>>(
        hid_bf, aw2t, ab2, nullptr, comb_f, gate, 0, /*store*/0, nullptr, F_DIM, 0);
    velora_gemm<<<dim3(F_DIM / 128, NTOK / 128), blk, 0, stream>>>(
        enh_bf, lw1t, lb1, hid_bf, nullptr, nullptr, 0, 0, nullptr, D_DIM, /*relu*/2);
    velora_gemm<<<dim3(D_DIM / 128, NTOK / 128), blk, 0, stream>>>(
        hid_bf, lw2t, lb2, nullptr, comb_f, gate, 1, /*accum*/1, nullptr, F_DIM, 0);

    // 5) fusion projection (writes both f32 for the residual and bf16 for ctx GEMM)
    velora_cvt_f32_bf16<<<cvt_grid, blk, 0, stream>>>(comb_f, comb_bf, (long)NTOK * D_DIM);
    velora_gemm<<<dim3(D_DIM / 128, NTOK / 128), blk, 0, stream>>>(
        comb_bf, fwt, fb, fused_bf, fused_f, nullptr, 0, 0, nullptr, D_DIM, 0);

    // 6) context manager + final residual average into d_out
    velora_gemm<<<dim3(D_DIM / 128, NTOK / 128), blk, 0, stream>>>(
        fused_bf, c1t, cb1, ctxh_bf, nullptr, nullptr, 0, 0, nullptr, D_DIM, /*tanh*/3);
    velora_gemm<<<dim3(D_DIM / 128, NTOK / 128), blk, 0, stream>>>(
        ctxh_bf, c2t, cb2, nullptr, out, nullptr, 0, 0, fused_f, D_DIM, 0);
}
